// MultiScaleAttention_90701119357702
// MI455X (gfx1250) — matrix operand order repaired and checked by reading, not compiled, not run on hardware
//
#include <hip/hip_runtime.h>
#include <math.h>

// ---------------------------------------------------------------------------
// MultiScaleAttention for MI455X (gfx1250): bf16 WMMA everywhere.
// B=1, S=2048, D=1024, H=16, DH=64, WIN=128, DIL=4
// Weights pre-transposed once so all WMMA B-fragments are two contiguous
// global_load_b128 per lane (k-contiguous rows), exploiting the 192MB L2
// instead of LDS staging.
// ---------------------------------------------------------------------------

#define S_LEN 2048
#define D_DIM 1024
#define H_NUM 16
#define DH_DIM 64

typedef __attribute__((ext_vector_type(16))) __bf16        v16bf;
typedef __attribute__((ext_vector_type(16))) unsigned short v16u;
typedef __attribute__((ext_vector_type(8)))  unsigned short v8u;
typedef __attribute__((ext_vector_type(8)))  float          v8f;

union F8  { v8f  v; float          f[8];  };
union U8  { v8u  v; unsigned short u[8];  };
union U16 { v16u v; unsigned short u[16]; };

__device__ __forceinline__ unsigned short f32_to_bf16(float f) {
  unsigned u = __builtin_bit_cast(unsigned, f);
  u += 0x7fffu + ((u >> 16) & 1u);          // round-to-nearest-even
  return (unsigned short)(u >> 16);
}

__device__ __forceinline__ v16bf pack_frag(v8u lo, v8u hi) {
  U8 a, b; a.v = lo; b.v = hi;
  U16 f;
#pragma unroll
  for (int e = 0; e < 8; ++e) { f.u[e] = a.u[e]; f.u[8 + e] = b.u[e]; }
  return __builtin_bit_cast(v16bf, f.v);
}

__device__ __forceinline__ float redmax16(float v) {   // reduce within 16-lane group
  v = fmaxf(v, __shfl_xor(v, 1, 32));
  v = fmaxf(v, __shfl_xor(v, 2, 32));
  v = fmaxf(v, __shfl_xor(v, 4, 32));
  v = fmaxf(v, __shfl_xor(v, 8, 32));
  return v;
}
__device__ __forceinline__ float redsum16(float v) {
  v += __shfl_xor(v, 1, 32);
  v += __shfl_xor(v, 2, 32);
  v += __shfl_xor(v, 4, 32);
  v += __shfl_xor(v, 8, 32);
  return v;
}

// ---------------------------------------------------------------------------
// Elementwise helpers (4-wide vectorized)
// ---------------------------------------------------------------------------
__global__ void cvt4_f32_bf16_kernel(const float* __restrict__ in,
                                     unsigned short* __restrict__ out, int n) {
  int i = (blockIdx.x * blockDim.x + threadIdx.x) * 4;
  if (i < n) {
    float4 f = *(const float4*)(in + i);
    union { unsigned short u[4]; unsigned long long ll; } o;
    o.u[0] = f32_to_bf16(f.x); o.u[1] = f32_to_bf16(f.y);
    o.u[2] = f32_to_bf16(f.z); o.u[3] = f32_to_bf16(f.w);
    *(unsigned long long*)(out + i) = o.ll;
  }
}

__global__ void zero4_f32_kernel(float* __restrict__ p, int n) {
  int i = (blockIdx.x * blockDim.x + threadIdx.x) * 4;
  if (i < n) *(float4*)(p + i) = make_float4(0.f, 0.f, 0.f, 0.f);
}

__global__ void scale_cvt4_kernel(const float* __restrict__ in,
                                  unsigned short* __restrict__ out, int n, float s) {
  int i = (blockIdx.x * blockDim.x + threadIdx.x) * 4;
  if (i < n) {
    float4 f = *(const float4*)(in + i);
    union { unsigned short u[4]; unsigned long long ll; } o;
    o.u[0] = f32_to_bf16(f.x * s); o.u[1] = f32_to_bf16(f.y * s);
    o.u[2] = f32_to_bf16(f.z * s); o.u[3] = f32_to_bf16(f.w * s);
    *(unsigned long long*)(out + i) = o.ll;
  }
}

// LDS-tiled transpose + fp32->bf16: in[K,N] -> out[N,K].  block (32,8), 32x32 tile
__global__ __launch_bounds__(256)
void cvt_transpose_kernel(const float* __restrict__ in,
                          unsigned short* __restrict__ out, int K, int N) {
  __shared__ float tile[32][33];
  const int k0 = blockIdx.y * 32, n0 = blockIdx.x * 32;
  const int tx = threadIdx.x, ty = threadIdx.y;
#pragma unroll
  for (int j = 0; j < 4; ++j)
    tile[ty + j * 8][tx] = in[(size_t)(k0 + ty + j * 8) * N + (n0 + tx)];
  __syncthreads();
#pragma unroll
  for (int j = 0; j < 4; ++j)
    out[(size_t)(n0 + ty + j * 8) * K + (k0 + tx)] = f32_to_bf16(tile[tx][ty + j * 8]);
}

// ---------------------------------------------------------------------------
// WMMA GEMM (TN): C[M,N] = A[M,K] * Bt[N,K]^T, bf16 in, fp32 accumulate.
// Block: 256 thr = 8 waves -> 128(M) x 64(N); all fragments direct from global
// (b128 pairs, L2-resident weights). No LDS, no barriers.
// mode 0: store fp32 into Cf[M,N]
// mode 1: QKV scatter (N = 3*D): q->Q[h][s][64], k->K[h][s][64], v->Vt[h][64][s]
// ---------------------------------------------------------------------------
__global__ __launch_bounds__(256)
void gemm_bf16_tn_wmma_kernel(const unsigned short* __restrict__ A,
                              const unsigned short* __restrict__ Bt,
                              float* __restrict__ Cf,
                              unsigned short* __restrict__ Qo,
                              unsigned short* __restrict__ Ko,
                              unsigned short* __restrict__ Vto,
                              int M, int N, int K, int mode) {
  const int tid  = threadIdx.x;
  const int wv   = tid >> 5;
  const int lane = tid & 31;
  const int lm   = lane & 15;
  const int half = lane >> 4;

  const int n0 = blockIdx.x * 64;
  const int m0 = blockIdx.y * 128 + wv * 16;

  const unsigned short* arow = A + (size_t)(m0 + lm) * K;
  const unsigned short* brow[4];
#pragma unroll
  for (int ns = 0; ns < 4; ++ns)
    brow[ns] = Bt + (size_t)(n0 + ns * 16 + lm) * K;

  F8 acc[4];
#pragma unroll
  for (int ns = 0; ns < 4; ++ns) acc[ns].v = (v8f){};

  for (int k0 = 0; k0 < K; k0 += 32) {
    // A fragment: k contiguous -> chunks at k0+half*16 and k0+half*16+8
    v16bf afrag = pack_frag(*(const v8u*)(arow + k0 + half * 16),
                            *(const v8u*)(arow + k0 + half * 16 + 8));
    if (k0 + 32 < K) __builtin_prefetch(arow + k0 + 32, 0, 1);   // global_prefetch_b8
#pragma unroll
    for (int ns = 0; ns < 4; ++ns) {
      // B fragment: k contiguous for fixed n -> two b128 from Bt row
      v16bf bfrag = pack_frag(*(const v8u*)(brow[ns] + k0 + half * 16),
                              *(const v8u*)(brow[ns] + k0 + half * 16 + 8));
      acc[ns].v = __builtin_amdgcn_wmma_f32_16x16x32_bf16(
          false, afrag, false, bfrag, (short)0, acc[ns].v, false, false);
    }
  }

  // Store C fragments: m = m0 + j + 8*half, n = n0 + ns*16 + lm
#pragma unroll
  for (int ns = 0; ns < 4; ++ns) {
    const int n = n0 + ns * 16 + lm;
#pragma unroll
    for (int j = 0; j < 8; ++j) {
      const int m = m0 + j + 8 * half;
      const float val = acc[ns].f[j];
      if (mode == 0) {
        Cf[(size_t)m * N + n] = val;
      } else {
        const int part = n >> 10;            // /D_DIM  (0=q 1=k 2=v)
        const int r    = n & (D_DIM - 1);
        const int h    = r >> 6;             // /DH_DIM
        const int dh   = r & (DH_DIM - 1);
        const unsigned short bv = f32_to_bf16(val);
        if (part == 0)      Qo [((size_t)h * M + m)  * DH_DIM + dh] = bv;
        else if (part == 1) Ko [((size_t)h * M + m)  * DH_DIM + dh] = bv;
        else                Vto[((size_t)h * DH_DIM + dh) * M + m ] = bv;
      }
    }
  }
}

// ---------------------------------------------------------------------------
// Flash-style WMMA attention over an index set:  s = sbase + idx*step
//   group g in [0,ngroups): sbase = g * sbase_mul, group length = glen
// One wave per 16-query tile; keys streamed in blocks of 32.
// qtl2 = log2(glen/16), ngl2 = log2(ngroups) (all powers of two here).
// Oacc[s][h*64+dh] += softmax(QK^T/sqrt(DH)) V   (plain +=, Oacc pre-zeroed)
// ---------------------------------------------------------------------------
__global__ __launch_bounds__(256)
void attn_wmma_kernel(const unsigned short* __restrict__ Q,
                      const unsigned short* __restrict__ Kb,
                      const unsigned short* __restrict__ Vt,
                      float* __restrict__ Oacc,
                      int qtl2, int ngl2, int glen, int step, int sbase_mul) {
  __shared__ unsigned short Pbuf[8][16][32];   // per-wave P tile, 8 KB

  const int wv    = threadIdx.x >> 5;
  const int lane  = threadIdx.x & 31;
  const int lm    = lane & 15;
  const int half  = lane >> 4;
  const int wave  = blockIdx.x * 8 + wv;

  const int qt  = wave & ((1 << qtl2) - 1);
  const int tmp = wave >> qtl2;
  const int g   = tmp & ((1 << ngl2) - 1);
  const int h   = tmp >> ngl2;
  const int sbase = g * sbase_mul;

  const unsigned short* Qh  = Q  + (size_t)h * S_LEN * DH_DIM;
  const unsigned short* Kh  = Kb + (size_t)h * S_LEN * DH_DIM;
  const unsigned short* Vth = Vt + (size_t)h * DH_DIM * S_LEN;

  const float scale = 0.125f;   // 1/sqrt(64)

  // Q fragments (A-layout), dh split into two 32-wide k-windows
  v16bf aq[2];
  {
    const int sq = sbase + (qt * 16 + lm) * step;
    const unsigned short* qrow = Qh + (size_t)sq * DH_DIM;
#pragma unroll
    for (int c = 0; c < 2; ++c)
      aq[c] = pack_frag(*(const v8u*)(qrow + c * 32 + half * 16),
                        *(const v8u*)(qrow + c * 32 + half * 16 + 8));
  }

  F8 o[4];
#pragma unroll
  for (int ns = 0; ns < 4; ++ns) o[ns].v = (v8f){};
  float mrun[8], lrun[8];
#pragma unroll
  for (int r = 0; r < 8; ++r) { mrun[r] = -__builtin_inff(); lrun[r] = 0.0f; }

  for (int kb = 0; kb < glen; kb += 32) {
    // prefetch next key block rows into cache (global_prefetch_b8)
    if (kb + 32 < glen) {
      __builtin_prefetch(Kh  + (size_t)(sbase + (kb + 32 + lm) * step) * DH_DIM, 0, 1);
      __builtin_prefetch(Vth + (size_t)lm * S_LEN + sbase + (kb + 32) * step, 0, 1);
    }

    // ---- scores: two 16-key sub-tiles ----
    F8 sfr[2];
#pragma unroll
    for (int u = 0; u < 2; ++u) {
      sfr[u].v = (v8f){};
      const int sk = sbase + (kb + u * 16 + lm) * step;
      const unsigned short* krow = Kh + (size_t)sk * DH_DIM;
#pragma unroll
      for (int c = 0; c < 2; ++c) {
        v16bf bk = pack_frag(*(const v8u*)(krow + c * 32 + half * 16),
                             *(const v8u*)(krow + c * 32 + half * 16 + 8));
        sfr[u].v = __builtin_amdgcn_wmma_f32_16x16x32_bf16(
            false, aq[c], false, bk, (short)0, sfr[u].v, false, false);
      }
#pragma unroll
      for (int r = 0; r < 8; ++r) sfr[u].f[r] *= scale;
    }

    // ---- online softmax (row m = r + 8*half lives on one 16-lane group) ----
#pragma unroll
    for (int r = 0; r < 8; ++r) {
      const float m_old = mrun[r];
      float rmax = redmax16(fmaxf(sfr[0].f[r], sfr[1].f[r]));
      const float mnew  = fmaxf(m_old, rmax);
      const float alpha = expf(m_old - mnew);
      const float p0 = expf(sfr[0].f[r] - mnew);
      const float p1 = expf(sfr[1].f[r] - mnew);
      const float rs = redsum16(p0 + p1);
      lrun[r] = lrun[r] * alpha + rs;
      mrun[r] = mnew;
#pragma unroll
      for (int ns = 0; ns < 4; ++ns) o[ns].f[r] *= alpha;
      Pbuf[wv][r + 8 * half][lm]      = f32_to_bf16(p0);
      Pbuf[wv][r + 8 * half][16 + lm] = f32_to_bf16(p1);
    }
    asm volatile("s_wait_dscnt 0" ::: "memory");   // LDS stores visible to loads

    // ---- P (A-layout from LDS) x V (B-layout from Vt rows) ----
    v16bf pfrag = pack_frag(*(const v8u*)&Pbuf[wv][lm][half * 16],
                            *(const v8u*)&Pbuf[wv][lm][half * 16 + 8]);
#pragma unroll
    for (int ns = 0; ns < 4; ++ns) {
      const unsigned short* vrow = Vth + (size_t)(ns * 16 + lm) * S_LEN;
      v16bf vfrag;
      if (step == 1) {
        const unsigned short* vb = vrow + sbase + kb + half * 16;
        vfrag = pack_frag(*(const v8u*)vb, *(const v8u*)(vb + 8));
      } else {
        U16 f;
#pragma unroll
        for (int e = 0; e < 8; ++e) {
          f.u[e]     = vrow[sbase + (kb + half * 16 + e) * step];
          f.u[8 + e] = vrow[sbase + (kb + half * 16 + 8 + e) * step];
        }
        vfrag = __builtin_bit_cast(v16bf, f.v);
      }
      o[ns].v = __builtin_amdgcn_wmma_f32_16x16x32_bf16(
          false, pfrag, false, vfrag, (short)0, o[ns].v, false, false);
    }
  }

  // ---- normalize and accumulate into Oacc[s][h*64+dh] ----
#pragma unroll
  for (int r = 0; r < 8; ++r) {
    const float inv = 1.0f / lrun[r];
    const int m  = r + 8 * half;
    const int sq = sbase + (qt * 16 + m) * step;
#pragma unroll
    for (int ns = 0; ns < 4; ++ns) {
      const int col = h * DH_DIM + ns * 16 + lm;
      Oacc[(size_t)sq * D_DIM + col] += o[ns].f[r] * inv;
    }
  }
}

// ---------------------------------------------------------------------------
// Host launcher
// ---------------------------------------------------------------------------
extern "C" void kernel_launch(void* const* d_in, const int* in_sizes, int n_in,
                              void* d_out, int out_size, void* d_ws, size_t ws_size,
                              hipStream_t stream) {
  const float* x      = (const float*)d_in[0];   // [S, D]
  const float* w_qkv  = (const float*)d_in[1];   // [D, 3D]
  const float* w_out  = (const float*)d_in[2];   // [D, D]
  float* out = (float*)d_out;                    // [S, D]

  const size_t nSD  = (size_t)S_LEN * D_DIM;       // 2,097,152
  const size_t nQKV = (size_t)D_DIM * 3 * D_DIM;   // 3,145,728
  const size_t nDD  = (size_t)D_DIM * D_DIM;       // 1,048,576

  char* ws = (char*)d_ws;
  size_t off = 0;
  auto alloc = [&](size_t bytes) { char* p = ws + off; off += (bytes + 255) & ~(size_t)255; return p; };

  unsigned short* xb     = (unsigned short*)alloc(nSD  * 2);
  unsigned short* wqkvT  = (unsigned short*)alloc(nQKV * 2);   // [3D, D] bf16
  unsigned short* woutT  = (unsigned short*)alloc(nDD  * 2);   // [D, D]  bf16
  unsigned short* Qb     = (unsigned short*)alloc(nSD  * 2);
  unsigned short* Kbuf   = (unsigned short*)alloc(nSD  * 2);
  unsigned short* Vtb    = (unsigned short*)alloc(nSD  * 2);
  float*          Oacc   = (float*)         alloc(nSD  * 4);
  unsigned short* Ob     = (unsigned short*)alloc(nSD  * 2);

  const int T = 256;
  // 1) x -> bf16; weights -> transposed bf16 [N,K]
  cvt4_f32_bf16_kernel<<<(int)((nSD / 4 + T - 1) / T), T, 0, stream>>>(x, xb, (int)nSD);
  {
    dim3 blk(32, 8);
    dim3 g1(3 * D_DIM / 32, D_DIM / 32);
    cvt_transpose_kernel<<<g1, blk, 0, stream>>>(w_qkv, wqkvT, D_DIM, 3 * D_DIM);
    dim3 g2(D_DIM / 32, D_DIM / 32);
    cvt_transpose_kernel<<<g2, blk, 0, stream>>>(w_out, woutT, D_DIM, D_DIM);
  }

  // 2) zero attention accumulator
  zero4_f32_kernel<<<(int)((nSD / 4 + T - 1) / T), T, 0, stream>>>(Oacc, (int)nSD);

  // 3) QKV projection GEMM (scatter into Q / K / Vt head-major layouts)
  {
    dim3 grid(3 * D_DIM / 64, S_LEN / 128);
    gemm_bf16_tn_wmma_kernel<<<grid, T, 0, stream>>>(xb, wqkvT, nullptr,
                                                     Qb, Kbuf, Vtb,
                                                     S_LEN, 3 * D_DIM, D_DIM, 1);
  }

  // 4) three attention scales, each adds into Oacc (2048 waves -> 256 blocks)
  //    (qtl2, ngl2, glen, step, sbase_mul)
  attn_wmma_kernel<<<256, T, 0, stream>>>(Qb, Kbuf, Vtb, Oacc, 3, 4,  128, 1, 128); // local
  attn_wmma_kernel<<<256, T, 0, stream>>>(Qb, Kbuf, Vtb, Oacc, 5, 2,  512, 4,   1); // dilated
  attn_wmma_kernel<<<256, T, 0, stream>>>(Qb, Kbuf, Vtb, Oacc, 7, 0, 2048, 1,   0); // global

  // 5) average of three scales -> bf16
  scale_cvt4_kernel<<<(int)((nSD / 4 + T - 1) / T), T, 0, stream>>>(Oacc, Ob, (int)nSD, 1.0f / 3.0f);

  // 6) output projection GEMM -> fp32 d_out
  {
    dim3 grid(D_DIM / 64, S_LEN / 128);
    gemm_bf16_tn_wmma_kernel<<<grid, T, 0, stream>>>(Ob, woutT, out,
                                                     nullptr, nullptr, nullptr,
                                                     S_LEN, D_DIM, D_DIM, 0);
  }
}
